// CNN_85383949844607
// MI455X (gfx1250) — matrix operand order repaired and checked by reading, not compiled, not run on hardware
//
#include <hip/hip_runtime.h>

typedef __attribute__((ext_vector_type(16))) _Float16 v16h;
typedef __attribute__((ext_vector_type(8)))  _Float16 v8h;
typedef __attribute__((ext_vector_type(4)))  _Float16 v4h;
typedef __attribute__((ext_vector_type(8)))  float    v8f;
typedef __attribute__((ext_vector_type(4)))  float    v4f;
typedef __attribute__((ext_vector_type(4)))  int      v4i;

#define SEQ 256
#define BB  32
#define EE  128
#define HH  128
#define FF  16
#define NCH 241
#define OUTD 5
#define G4  512   // 4*H
#define WMAT (G4*EE)   // 65536 halves per weight matrix

#if __has_builtin(__builtin_amdgcn_global_load_async_to_lds_b128) && \
    __has_builtin(__builtin_amdgcn_s_wait_asynccnt)
#define ASYNC_LDS 1
#endif

__device__ __forceinline__ float fsigmoid(float x){ return 1.0f/(1.0f + __expf(-x)); }
__device__ __forceinline__ float ftanh(float x){ return 1.0f - 2.0f/(__expf(2.0f*x)+1.0f); }

__device__ __forceinline__ v16h cat8(v8h lo, v8h hi2){
  v16h a;
  #pragma unroll
  for (int j=0;j<8;++j){ a[j]=lo[j]; a[j+8]=hi2[j]; }
  return a;
}

// Stage n_halves f16 values from global into LDS, 16B per lane per issue.
__device__ __forceinline__ void lds_fill(_Float16* dst, const _Float16* src,
                                         int n_halves, int tid, int nthr){
#ifdef ASYNC_LDS
  for (int i = tid*8; i < n_halves; i += nthr*8){
    __builtin_amdgcn_global_load_async_to_lds_b128(
        (__attribute__((address_space(1))) v4i*)(void*)(src + i),
        (__attribute__((address_space(3))) v4i*)(void*)(dst + i),
        0, 0);
  }
  __builtin_amdgcn_s_wait_asynccnt(0);
#else
  const v8h* s8 = (const v8h*)src;
  v8h*       d8 = (v8h*)dst;
  for (int i = tid; i < n_halves/8; i += nthr) d8[i] = s8[i];
#endif
}

// ---------------------------------------------------------------------------
// Kernel 0: convert the four weight matrices [512x128] f32 -> f16 in ws.
// order: 0=w_ih_f, 1=w_ih_b, 2=w_hh_f, 3=w_hh_b
// ---------------------------------------------------------------------------
__global__ __launch_bounds__(256) void k_cvt_w(
    const float* __restrict__ w_ih_f, const float* __restrict__ w_ih_b,
    const float* __restrict__ w_hh_f, const float* __restrict__ w_hh_b,
    _Float16* __restrict__ w16)
{
  const int m = blockIdx.y;
  const float* src = (m==0) ? w_ih_f : (m==1) ? w_ih_b : (m==2) ? w_hh_f : w_hh_b;
  const int i = blockIdx.x*256 + threadIdx.x;      // 0..16383 float4s
  const float4* s4 = (const float4*)src;
  v4h* d4 = (v4h*)(w16 + (size_t)m*WMAT);
  float4 v = s4[i];
  v4h h; h[0]=(_Float16)v.x; h[1]=(_Float16)v.y; h[2]=(_Float16)v.z; h[3]=(_Float16)v.w;
  d4[i] = h;
}

// ---------------------------------------------------------------------------
// Kernel 1: xg[b*S+s][4H] = emb[text[s,b]] @ w_ih.T + bias   (both directions)
// WMMA: A = w_ih tile (16 gate cols x 32 K), B = emb rows (32 K x 16 cells)
// grid: 256 blocks x 128 thr. blocks [0,128) -> fwd, [128,256) -> bwd.
// ---------------------------------------------------------------------------
__global__ __launch_bounds__(128) void k_embed_gates(
    const int* __restrict__ text, const float* __restrict__ emb,
    const _Float16* __restrict__ w16,
    const float* __restrict__ b_f, const float* __restrict__ b_b,
    float* __restrict__ xg_f, float* __restrict__ xg_b)
{
  __shared__ __align__(16) _Float16 w_lds[WMAT];
  __shared__ __align__(16) float    bias_lds[G4];

  const int tid = threadIdx.x;
  const int dir = blockIdx.x >> 7;
  const int blk = blockIdx.x & 127;
  const float* bs = dir ? b_b : b_f;
  float* xg = dir ? xg_b : xg_f;

  lds_fill(w_lds, w16 + (size_t)dir*WMAT, WMAT, tid, 128);
  for (int i = tid; i < G4; i += 128) bias_lds[i] = bs[i];
  __syncthreads();

  const int wave = tid >> 5;
  const int lane = tid & 31;
  const int ln   = lane & 15;
  const int hi   = lane >> 4;
  const int group = blk*4 + wave;          // 0..511 groups of 16 rows
  const int row   = group*16 + ln;         // row = b*SEQ + s
  const int bb = row >> 8;
  const int ss = row & 255;
  const int tok = text[ss*BB + bb];

  // B fragments: gathered embedding row, f32 -> f16
  v16h bfrag[4];
  #pragma unroll
  for (int kk=0;kk<4;++kk){
    const float* src = emb + (size_t)tok*EE + kk*32 + hi*8;
    v4f f0 = *(const v4f*)(src);
    v4f f1 = *(const v4f*)(src+4);
    v4f f2 = *(const v4f*)(src+16);
    v4f f3 = *(const v4f*)(src+20);
    v16h x;
    #pragma unroll
    for (int j=0;j<4;++j){
      x[j]=(_Float16)f0[j]; x[4+j]=(_Float16)f1[j];
      x[8+j]=(_Float16)f2[j]; x[12+j]=(_Float16)f3[j];
    }
    bfrag[kk]=x;
  }

  for (int m16 = 0; m16 < 32; ++m16){
    v8f acc = {};
    #pragma unroll
    for (int kk=0;kk<4;++kk){
      int off = (m16*16 + ln)*EE + kk*32 + hi*8;
      v16h af = cat8(*(const v8h*)(&w_lds[off]), *(const v8h*)(&w_lds[off+16]));
      acc = __builtin_amdgcn_wmma_f32_16x16x32_f16(false, af, false, bfrag[kk],
                                                   (short)0, acc, false, false);
    }
    const int gb = m16*16 + hi*8;
    v4f b0 = *(const v4f*)(&bias_lds[gb]);
    v4f b1 = *(const v4f*)(&bias_lds[gb+4]);
    v4f o0, o1;
    #pragma unroll
    for (int j=0;j<4;++j){ o0[j]=acc[j]+b0[j]; o1[j]=acc[4+j]+b1[j]; }
    float* dst = xg + (size_t)row*G4 + gb;
    *(v4f*)(dst)   = o0;
    *(v4f*)(dst+4) = o1;
  }
}

// ---------------------------------------------------------------------------
// Kernel 2: masked LSTM recurrence over 16 steps, both directions.
// Per wave: 16 cells = (batch bb, chunks c0..c0+15). WMMA D = W_hh x h^T.
// h kept as f16 in per-wave LDS tile; conditional store = packed-seq "hold".
// grid: 256 blocks x 128 thr. blocks [0,128) -> fwd, [128,256) -> bwd.
// ---------------------------------------------------------------------------
__global__ __launch_bounds__(128) void k_lstm(
    const int* __restrict__ text_lengths,
    const _Float16* __restrict__ w16,
    const float* __restrict__ xg_f, const float* __restrict__ xg_b,
    float* __restrict__ hid)
{
  __shared__ __align__(16) _Float16 w_lds[WMAT];
  __shared__ __align__(16) _Float16 h_stage[4][16*HH];

  const int tid = threadIdx.x;
  const int dir = blockIdx.x >> 7;
  const int blk = blockIdx.x & 127;
  const float* xg = dir ? xg_b : xg_f;

  lds_fill(w_lds, w16 + (size_t)(2+dir)*WMAT, WMAT, tid, 128);

  const int wave = tid >> 5;
  const int lane = tid & 31;
  const int ln   = lane & 15;
  const int hi   = lane >> 4;
  const int tile = blk*4 + wave;       // 0..511
  const int bb   = tile >> 4;
  const int c0   = (tile & 15) * 16;
  const int c    = c0 + ln;            // this lane's chunk (cell)
  const int len_b = text_lengths[bb];
  int lens = len_b - c; lens = lens < 1 ? 1 : (lens > FF ? FF : lens);

  _Float16* hl = &h_stage[wave][0];
  {
    v8h z = {};
    #pragma unroll
    for (int j=0;j<8;++j) *(v8h*)(hl + ln*HH + hi*64 + j*8) = z;
  }
  __syncthreads();

  v8f cst[8];
  #pragma unroll
  for (int n=0;n<8;++n) cst[n] = (v8f){};

  for (int t = 0; t < FF; ++t){
    int tt = dir ? (lens-1-t < 0 ? 0 : lens-1-t) : t;   // bwd: reversed valid prefix
    int srow = c + tt; if (srow > SEQ-1) srow = SEQ-1;
    const float* xr = xg + (size_t)(bb*SEQ + srow)*G4;

    // prefetch next step's xg row (lanes fan out across the 2KB row)
    if (t + 1 < FF){
      int t2 = t + 1;
      int tn = dir ? (lens-1-t2 < 0 ? 0 : lens-1-t2) : t2;
      int srp = c + tn; if (srp > SEQ-1) srp = SEQ-1;
      const float* xp = xg + (size_t)(bb*SEQ + srp)*G4;
      __builtin_prefetch(xp + lane*16, 0, 1);
    }

    // B fragments from current h (f16, per-wave LDS tile)
    v16h hb[4];
    #pragma unroll
    for (int kk=0;kk<4;++kk){
      int off = ln*HH + kk*32 + hi*8;
      hb[kk] = cat8(*(const v8h*)(hl+off), *(const v8h*)(hl+off+16));
    }

    const bool upd = (t < lens);

    #pragma unroll
    for (int n=0;n<8;++n){            // 16-wide slices of H
      v8f acc[4];
      #pragma unroll
      for (int g=0; g<4; ++g){        // i, f, g, o
        const int m16 = g*8 + n;
        const int gb  = m16*16 + hi*8;
        v4f x0 = *(const v4f*)(xr + gb);
        v4f x1 = *(const v4f*)(xr + gb + 4);
        v8f a;
        #pragma unroll
        for (int j=0;j<4;++j){ a[j]=x0[j]; a[4+j]=x1[j]; }
        #pragma unroll
        for (int kk=0;kk<4;++kk){
          int off = (m16*16 + ln)*HH + kk*32 + hi*8;
          v16h af = cat8(*(const v8h*)(&w_lds[off]), *(const v8h*)(&w_lds[off+16]));
          a = __builtin_amdgcn_wmma_f32_16x16x32_f16(false, af, false, hb[kk],
                                                     (short)0, a, false, false);
        }
        acc[g]=a;
      }
      v8h hn;
      #pragma unroll
      for (int r=0;r<8;++r){
        float ig = fsigmoid(acc[0][r]);
        float fg = fsigmoid(acc[1][r]);
        float gg = ftanh  (acc[2][r]);
        float og = fsigmoid(acc[3][r]);
        float cn = fg*cst[n][r] + ig*gg;
        if (upd) cst[n][r] = cn;
        hn[r] = (_Float16)(og*ftanh(cn));
      }
      if (upd) *(v8h*)(hl + ln*HH + n*16 + hi*8) = hn;   // hold carry when !upd
    }
    __syncthreads();
  }

  // final hidden -> hid[c][b][dir*H + k] (f32)
  #pragma unroll
  for (int j=0;j<8;++j){
    const int hc = hi*64 + j*8;
    v8h hv = *(const v8h*)(hl + ln*HH + hc);
    if (c < NCH){
      float* dst = hid + ((size_t)c*BB + bb)*(2*HH) + dir*HH + hc;
      v4f o0,o1;
      #pragma unroll
      for (int jj=0;jj<4;++jj){ o0[jj]=(float)hv[jj]; o1[jj]=(float)hv[4+jj]; }
      *(v4f*)dst     = o0;
      *(v4f*)(dst+4) = o1;
    }
  }
}

// ---------------------------------------------------------------------------
// Kernel 3: max over chunks + FC.  One block per batch row.
// ---------------------------------------------------------------------------
__global__ __launch_bounds__(256) void k_pool_fc(
    const float* __restrict__ hid, const float* __restrict__ w_fc,
    const float* __restrict__ b_fc, float* __restrict__ out)
{
  __shared__ float pooled[2*HH];
  const int bb = blockIdx.x;
  const int k  = threadIdx.x;
  float m = -3.4e38f;
  for (int cc = 0; cc < NCH; ++cc){
    float v = hid[((size_t)cc*BB + bb)*(2*HH) + k];
    m = v > m ? v : m;
  }
  pooled[k] = m;
  __syncthreads();
  if (k < OUTD){
    float acc = b_fc[k];
    #pragma unroll 8
    for (int j=0;j<2*HH;++j) acc += pooled[j]*w_fc[k*2*HH + j];
    out[bb*OUTD + k] = acc;
  }
}

// ---------------------------------------------------------------------------
extern "C" void kernel_launch(void* const* d_in, const int* in_sizes, int n_in,
                              void* d_out, int out_size, void* d_ws, size_t ws_size,
                              hipStream_t stream) {
  const int*   text         = (const int*)  d_in[0];
  const int*   text_lengths = (const int*)  d_in[1];
  const float* emb          = (const float*)d_in[2];
  const float* w_ih_f       = (const float*)d_in[3];
  const float* w_hh_f       = (const float*)d_in[4];
  const float* b_f          = (const float*)d_in[5];
  const float* w_ih_b       = (const float*)d_in[6];
  const float* w_hh_b       = (const float*)d_in[7];
  const float* b_b          = (const float*)d_in[8];
  const float* w_fc         = (const float*)d_in[9];
  const float* b_fc         = (const float*)d_in[10];
  float* out = (float*)d_out;

  float* xg_f = (float*)d_ws;                       // [8192][512]
  float* xg_b = xg_f + (size_t)BB*SEQ*G4;           // [8192][512]
  float* hid  = xg_b + (size_t)BB*SEQ*G4;           // [241][32][256]
  _Float16* w16 = (_Float16*)(hid + (size_t)NCH*BB*2*HH);  // [4][65536] f16

  k_cvt_w<<<dim3(64,4), 256, 0, stream>>>(w_ih_f, w_ih_b, w_hh_f, w_hh_b, w16);
  k_embed_gates<<<256, 128, 0, stream>>>(text, emb, w16, b_f, b_b, xg_f, xg_b);
  k_lstm<<<256, 128, 0, stream>>>(text_lengths, w16, xg_f, xg_b, hid);
  k_pool_fc<<<BB, 256, 0, stream>>>(hid, w_fc, b_fc, out);
}
